// LocationSensitiveAttention_74148315398764
// MI455X (gfx1250) — matrix operand order repaired and checked by reading, not compiled, not run on hardware
//
#include <hip/hip_runtime.h>
#include <hip/hip_bf16.h>

typedef __attribute__((ext_vector_type(16))) __bf16 v16bf;
typedef __attribute__((ext_vector_type(8)))  float  v8f;

#define Bsz 128
#define Tsz 1024
#define Dsz 512
#define Qsz 1024
#define Asz 128
#define Ksz 31
#define PAD 15
#define TM  64          // t-rows per energy block
#define NEG_MAX (-3.4028234663852886e38f)

// ---------------------------------------------------------------------------
// Kernel 1: pq = query[B,Q] @ W_query[Q,A]  -> pq[B,A]   (WMMA bf16)
// one wave per 16x16 output tile, grid = (B/16)*(A/16) = 64 waves
// ---------------------------------------------------------------------------
__global__ __launch_bounds__(32) void pq_kernel(const float* __restrict__ query,
                                                const float* __restrict__ Wq,
                                                float* __restrict__ pq) {
  const int lane = threadIdx.x;
  const int bm = blockIdx.x >> 3;          // M tile over B
  const int bn = blockIdx.x & 7;           // N tile over A
  const int half = (lane >= 16);
  const int l15  = lane & 15;
  const int rowA = bm * 16 + l15;
  const int colB = bn * 16 + l15;
  const int koffA  = half ? 8  : 0;        // A frag: K = {koff..koff+7} U {16+koff..}
  const int kbaseB = half ? 8 : 0;         // B frag: K = {kbase..kbase+7} U {16+kbase..}

  v8f acc = {};
  for (int k0 = 0; k0 < Qsz; k0 += 32) {
    v16bf a, b;
    const float* qrow = query + (size_t)rowA * Qsz + k0;
#pragma unroll
    for (int i = 0; i < 8; ++i) {
      a[i]     = (__bf16)qrow[koffA + i];
      a[8 + i] = (__bf16)qrow[16 + koffA + i];
    }
#pragma unroll
    for (int i = 0; i < 16; ++i)
      b[i] = (__bf16)Wq[(size_t)(k0 + kbaseB + i + (i >= 8 ? 8 : 0)) * Asz + colB];
    acc = __builtin_amdgcn_wmma_f32_16x16x32_bf16(false, a, false, b,
                                                  (short)0, acc, false, false);
  }
#pragma unroll
  for (int r = 0; r < 8; ++r) {
    int row = bm * 16 + r + (half ? 8 : 0);
    pq[row * Asz + colB] = acc[r];
  }
}

// ---------------------------------------------------------------------------
// Kernel 2: WK[k,c,a] = sum_f conv_kernel[k,c,f] * W_loc[f,a]   (tiny fold)
// ---------------------------------------------------------------------------
__global__ __launch_bounds__(256) void wk_kernel(const float* __restrict__ ck,    // [31,2,32]
                                                 const float* __restrict__ Wloc,  // [32,128]
                                                 float* __restrict__ WK) {        // [31,2,128]
  const int k = blockIdx.x;
  const int c = threadIdx.x >> 7;
  const int a = threadIdx.x & 127;
  float s = 0.f;
#pragma unroll
  for (int f = 0; f < 32; ++f)
    s += ck[(k * 2 + c) * 32 + f] * Wloc[f * Asz + a];
  WK[(k * 2 + c) * Asz + a] = s;
}

// ---------------------------------------------------------------------------
// Kernel 3: fused ploc-GEMM (WMMA) + keys-GEMM (WMMA, pipelined) + energy
// grid = B*(T/TM), 256 threads (8 waves). Wave w owns N-tile w of A=128.
// acc starts as the ploc tile (win[64x64] @ WKb[64x128], 2 k-steps), then the
// keys GEMM (memory @ W_memory, 16 k-steps) accumulates on top.
// ---------------------------------------------------------------------------
__global__ __launch_bounds__(256) void energy_kernel(
    const float* __restrict__ memory,   // [B,T,D]
    const float* __restrict__ Wmem,     // [D,A]
    const float* __restrict__ pq,       // [B,A]     (ws)
    const float* __restrict__ WK,       // [31,2,A]  (ws)
    const float* __restrict__ prevw,    // [B,T]
    const float* __restrict__ cumw,     // [B,T]
    const float* __restrict__ vvec,     // [A]
    const int*   __restrict__ seqlen,   // [B]
    float* __restrict__ score)          // [B,T]     (ws)
{
  __shared__ __bf16 ldsA[TM * 32];       // keys A tile (64 x 32)
  __shared__ __bf16 winA[TM * 64];       // conv window matrix (64 x 64, K-padded)
  __shared__ __bf16 wkB [64 * Asz];      // folded conv weights (64 x 128, K-padded)
  __shared__ float  scoreAcc[TM];

  const int b   = blockIdx.x >> 4;
  const int t0  = (blockIdx.x & 15) * TM;
  const int tid  = threadIdx.x;
  const int wave = tid >> 5;
  const int lane = tid & 31;
  const int half = (lane >= 16);
  const int l15  = lane & 15;
  const int koffA  = half ? 8  : 0;
  const int kbaseB = half ? 8 : 0;
  const int colN = wave * 16 + l15;      // this lane's attention-dim index

  // --- stage win matrix: win[t][j] = j<31 ? prev[t0+t+j-15] : cum[t0+t+(j-32)-15]
  for (int idx = tid; idx < TM * 64; idx += 256) {
    int t = idx >> 6, j = idx & 63;
    float v = 0.f;
    if (j != 31 && j != 63) {
      int k = (j < 31) ? j : (j - 32);
      const float* src = (j < 31) ? prevw : cumw;
      int tg = t0 + t + k - PAD;
      if (tg >= 0 && tg < Tsz) v = src[b * Tsz + tg];
    }
    winA[idx] = (__bf16)v;
  }
  // --- stage WKb: wkB[j][a] = j<31 ? WK[j,0,a] : WK[j-32,1,a]  (rows 31,63 = 0)
  for (int idx = tid; idx < 64 * Asz; idx += 256) {
    int j = idx >> 7, a = idx & 127;
    float v = 0.f;
    if (j != 31 && j != 63)
      v = (j < 31) ? WK[(j * 2) * Asz + a] : WK[((j - 32) * 2 + 1) * Asz + a];
    wkB[idx] = (__bf16)v;
  }
  if (tid < TM) scoreAcc[tid] = 0.f;

  // start first keys tile loads early (overlap with ploc WMMAs below)
  const size_t memBase = ((size_t)b * Tsz + t0) * Dsz;
  float regs[8];
#pragma unroll
  for (int j = 0; j < 8; ++j) {
    int idx = tid + j * 256;
    regs[j] = memory[memBase + (size_t)(idx >> 5) * Dsz + (idx & 31)];
  }
  __syncthreads();

  v8f acc[4] = {};

  // --- ploc GEMM: acc = win @ wkB   (2 k-steps of 32)
#pragma unroll
  for (int ks = 0; ks < 2; ++ks) {
    v16bf bf;
#pragma unroll
    for (int i = 0; i < 16; ++i)
      bf[i] = wkB[(ks * 32 + kbaseB + i + (i >= 8 ? 8 : 0)) * Asz + colN];
#pragma unroll
    for (int mt = 0; mt < 4; ++mt) {
      v16bf af;
      const __bf16* ap = winA + (mt * 16 + l15) * 64 + ks * 32;
#pragma unroll
      for (int i = 0; i < 8; ++i) {
        af[i]     = ap[koffA + i];
        af[8 + i] = ap[16 + koffA + i];
      }
      acc[mt] = __builtin_amdgcn_wmma_f32_16x16x32_bf16(false, af, false, bf,
                                                        (short)0, acc[mt], false, false);
    }
  }

  // --- keys GEMM: acc += memory_tile @ Wmem  (16 k-steps, software pipelined)
  for (int k0 = 0; k0 < Dsz; k0 += 32) {
#pragma unroll
    for (int j = 0; j < 8; ++j)
      ldsA[tid + j * 256] = (__bf16)regs[j];
    __syncthreads();

    if (k0 + 32 < Dsz) {           // next tile -> registers (in flight during WMMA)
#pragma unroll
      for (int j = 0; j < 8; ++j) {
        int idx = tid + j * 256;
        regs[j] = memory[memBase + (size_t)(idx >> 5) * Dsz + k0 + 32 + (idx & 31)];
      }
    }
    if (k0 + 64 < Dsz) {           // prefetch two tiles ahead (global_prefetch_b8)
      int row = tid >> 2, c = (tid & 3) * 8;
      __builtin_prefetch(&memory[memBase + (size_t)row * Dsz + k0 + 64 + c], 0, 1);
    }

    v16bf bf;
#pragma unroll
    for (int i = 0; i < 16; ++i)
      bf[i] = (__bf16)Wmem[(size_t)(k0 + kbaseB + i + (i >= 8 ? 8 : 0)) * Asz + colN];
#pragma unroll
    for (int mt = 0; mt < 4; ++mt) {
      v16bf af;
      const __bf16* ap = ldsA + (mt * 16 + l15) * 32;
#pragma unroll
      for (int i = 0; i < 8; ++i) {
        af[i]     = ap[koffA + i];
        af[8 + i] = ap[16 + koffA + i];
      }
      acc[mt] = __builtin_amdgcn_wmma_f32_16x16x32_bf16(false, af, false, bf,
                                                        (short)0, acc[mt], false, false);
    }
    __syncthreads();
  }

  // --- energy: e = tanh(pq + ploc + keys) * v[a]; reduce over a
  const float pqv = pq[b * Asz + colN];
  const float vv  = vvec[colN];
#pragma unroll
  for (int mt = 0; mt < 4; ++mt) {
#pragma unroll
    for (int r = 0; r < 8; ++r) {
      int m = mt * 16 + r + (half ? 8 : 0);
      float e = tanhf(pqv + acc[mt][r]) * vv;
      e += __shfl_xor(e, 1, 16);           // reduce the 16 lanes sharing row m
      e += __shfl_xor(e, 2, 16);
      e += __shfl_xor(e, 4, 16);
      e += __shfl_xor(e, 8, 16);
      if (l15 == 0) atomicAdd(&scoreAcc[m], e);   // 8 waves x 1 half per row
    }
  }
  __syncthreads();

  if (tid < TM) {
    int t = t0 + tid;
    float s = (t < seqlen[b]) ? scoreAcc[tid] : NEG_MAX;
    score[b * Tsz + t] = s;
  }
}

// ---------------------------------------------------------------------------
// Kernel 4: masked softmax over T + new_cum. One block per batch row.
// ---------------------------------------------------------------------------
__global__ __launch_bounds__(256) void softmax_kernel(
    const float* __restrict__ score,    // [B,T] (ws)
    const float* __restrict__ cumw,     // [B,T]
    float* __restrict__ attn,           // [B,T]
    float* __restrict__ newcum)         // [B,T]
{
  __shared__ float w[Tsz];
  __shared__ float red[256];
  const int b = blockIdx.x, tid = threadIdx.x;

  float mx = NEG_MAX;
  for (int t = tid; t < Tsz; t += 256) {
    float s = score[b * Tsz + t];
    w[t] = s;
    mx = fmaxf(mx, s);
  }
  red[tid] = mx; __syncthreads();
  for (int s = 128; s > 0; s >>= 1) {
    if (tid < s) red[tid] = fmaxf(red[tid], red[tid + s]);
    __syncthreads();
  }
  mx = red[0]; __syncthreads();

  float sum = 0.f;
  for (int t = tid; t < Tsz; t += 256) {
    float e = __expf(w[t] - mx);
    w[t] = e;
    sum += e;
  }
  red[tid] = sum; __syncthreads();
  for (int s = 128; s > 0; s >>= 1) {
    if (tid < s) red[tid] += red[tid + s];
    __syncthreads();
  }
  const float inv = 1.f / red[0];

  for (int t = tid; t < Tsz; t += 256) {
    float ww = w[t] * inv;
    attn[b * Tsz + t]   = ww;
    newcum[b * Tsz + t] = ww + cumw[b * Tsz + t];
  }
}

// ---------------------------------------------------------------------------
// Kernel 5: context[b,d] = sum_t attn[b,t]*memory[b,t,d]
// grid = B*4 blocks x 128 threads: block handles 128 d-columns, coalesced.
// ---------------------------------------------------------------------------
__global__ __launch_bounds__(128) void context_kernel(
    const float* __restrict__ attn,     // [B,T]
    const float* __restrict__ memory,   // [B,T,D]
    float* __restrict__ context)        // [B,D]
{
  __shared__ float w[Tsz];
  const int b  = blockIdx.x >> 2;
  const int d  = (blockIdx.x & 3) * 128 + threadIdx.x;
  for (int t = threadIdx.x; t < Tsz; t += 128)
    w[t] = attn[b * Tsz + t];
  __syncthreads();

  const size_t memBase = (size_t)b * Tsz * Dsz + d;
  float acc = 0.f;
  for (int t = 0; t < Tsz; ++t)
    acc += w[t] * memory[memBase + (size_t)t * Dsz];
  context[b * Dsz + d] = acc;
}

// ---------------------------------------------------------------------------
extern "C" void kernel_launch(void* const* d_in, const int* in_sizes, int n_in,
                              void* d_out, int out_size, void* d_ws, size_t ws_size,
                              hipStream_t stream) {
  const float* query  = (const float*)d_in[0];
  const float* prevw  = (const float*)d_in[1];
  const float* cumw   = (const float*)d_in[2];
  const float* memory = (const float*)d_in[3];
  const int*   seqlen = (const int*)  d_in[4];
  const float* Wq     = (const float*)d_in[5];
  const float* Wm     = (const float*)d_in[6];
  const float* ck     = (const float*)d_in[7];
  const float* Wloc   = (const float*)d_in[8];
  const float* vvec   = (const float*)d_in[9];

  float* out     = (float*)d_out;
  float* context = out;                          // [B,D]
  float* attn    = context + Bsz * Dsz;          // [B,T]
  float* newcum  = attn + Bsz * Tsz;             // [B,T]

  float* ws    = (float*)d_ws;
  float* pq    = ws;                             // [B,A]
  float* WK    = pq + Bsz * Asz;                 // [31,2,A]
  float* score = WK + Ksz * 2 * Asz;             // [B,T]

  pq_kernel<<<(Bsz / 16) * (Asz / 16), 32, 0, stream>>>(query, Wq, pq);
  wk_kernel<<<Ksz, 256, 0, stream>>>(ck, Wloc, WK);
  energy_kernel<<<Bsz * (Tsz / TM), 256, 0, stream>>>(memory, Wm, pq, WK,
                                                      prevw, cumw, vvec, seqlen, score);
  softmax_kernel<<<Bsz, 256, 0, stream>>>(score, cumw, attn, newcum);
  context_kernel<<<Bsz * 4, 128, 0, stream>>>(attn, memory, context);
}
